// LaplaceCostNet_60301340836008
// MI455X (gfx1250) — compile-verified, matrix operand order checked by reading, not run on hardware
//
#include <hip/hip_runtime.h>

typedef __attribute__((ext_vector_type(16))) _Float16 v16h;
typedef __attribute__((ext_vector_type(8)))  _Float16 v8h;
typedef __attribute__((ext_vector_type(8)))  float    v8f;

#define RH 512
#define RW 512
#define NPIX (RH * RW)
#define PITCH 136      // halfs per LDS row (128 + 8 pad -> 272B, 16B aligned, bank-skewed)
#define MTILE 128      // points per workgroup (4 waves x 32 rows)
#define MAXV 100.0f

// ---------------------------------------------------------------------------
// Fragment load matching CDNA5 16-bit A/B WMMA layout:
//   lanes 0-15 : K = kb..kb+7 (VGPR0-3), kb+16..kb+23 (VGPR4-7)
//   lanes 16-31: K = kb+8..kb+15,        kb+24..kb+31   (handled via kbase+8)
// ---------------------------------------------------------------------------
__device__ __forceinline__ v16h load_frag(const _Float16* rowbase, int kbase) {
    v8h lo = *(const v8h*)(rowbase + kbase);
    v8h hi = *(const v8h*)(rowbase + kbase + 16);
    return __builtin_shufflevector(lo, hi, 0,1,2,3,4,5,6,7,8,9,10,11,12,13,14,15);
}

// ---------------------------------------------------------------------------
// K0: V0 init + obstacle mask
// ---------------------------------------------------------------------------
__global__ void init_kernel(const int* __restrict__ bt, const float* __restrict__ bc,
                            float* __restrict__ V0, float* __restrict__ obj) {
    int idx = blockIdx.x * blockDim.x + threadIdx.x;
    if (idx >= NPIX) return;
    int   t = bt[idx];
    float b = bc[idx];
    V0[idx]  = (t == 0) ? MAXV : b;
    obj[idx] = (t == 1 && b > 0.f) ? b : 0.f;
}

// ---------------------------------------------------------------------------
// K1: weight convert f32 [l][k][n] -> f16 transposed [l][n][k]
// ---------------------------------------------------------------------------
__global__ void wconv_kernel(const float* __restrict__ Wh, _Float16* __restrict__ Wt16) {
    int idx = blockIdx.x * blockDim.x + threadIdx.x;   // 4*128*128 threads
    int l   = idx >> 14;
    int rem = idx & 16383;
    int k   = rem >> 7;
    int n   = rem & 127;
    Wt16[l * 16384 + n * 128 + k] = (_Float16)Wh[l * 16384 + k * 128 + n];
}

// ---------------------------------------------------------------------------
// K2: fused Fourier encoding + 4x(128x128) WMMA MLP + output head
// 128 threads = 4 wave32 = 128 points. Wave w owns rows 32w..32w+31
// (two 16-row WMMA tiles) so each B fragment feeds two WMMAs.
// ---------------------------------------------------------------------------
__global__ void mlp_kernel(const float* __restrict__ x,
                           const float* __restrict__ Benc,
                           const _Float16* __restrict__ Wt16,
                           const float* __restrict__ Wout,
                           float* __restrict__ C_out,
                           float* __restrict__ Cpos) {
    extern __shared__ __align__(16) _Float16 smem[];
    _Float16* actA = smem;                       // 128 x 136 halfs
    _Float16* actB = actA + MTILE * PITCH;       // 128 x 136 halfs
    _Float16* wbuf = actB + MTILE * PITCH;       // 128 x 136 halfs (current layer, [n][k])

    const int tid  = threadIdx.x;
    const int wg   = blockIdx.x;
    const int lane = tid & 31;
    const int wave = tid >> 5;

    // ---- Gaussian Fourier feature encoding: 1 point per thread ----
    {
        int gp = wg * MTILE + tid;
        float x0 = 0.5f + 0.5f * x[gp * 2 + 0];
        float x1 = 0.5f + 0.5f * x[gp * 2 + 1];
        const float TWO_PI = 6.283185307179586f;
        #pragma unroll
        for (int f = 0; f < 64; ++f) {
            float vp = TWO_PI * (x0 * Benc[f * 2 + 0] + x1 * Benc[f * 2 + 1]);
            float s, c;
            __sincosf(vp, &s, &c);
            actA[tid * PITCH + f]      = (_Float16)c;   // cos block: cols 0..63
            actA[tid * PITCH + 64 + f] = (_Float16)s;   // sin block: cols 64..127
        }
    }
    __syncthreads();

    _Float16* src = actA;
    _Float16* dst = actB;
    const int r0      = wave * 32;
    const int hi_half = (lane >> 4) ? 8 : 0;     // K-offset for lanes 16..31
    const int mhalf   = (lane >> 4) ? 8 : 0;     // M-offset of C/D rows for lanes 16..31
    const int nlane   = lane & 15;

    for (int layer = 0; layer < 4; ++layer) {
        // cooperative stage of this layer's transposed weights into LDS
        const _Float16* wgl = Wt16 + layer * 16384;
        for (int i = tid; i < 16384 / 8; i += 128) {
            int e    = i * 8;
            int nrow = e >> 7;
            int kcol = e & 127;
            *(v8h*)(wbuf + nrow * PITCH + kcol) = *(const v8h*)(wgl + e);
        }
        __syncthreads();

        // preload this wave's 8 A fragments (two 16-row tiles, K=0..127)
        v16h afrag[2][4];
        #pragma unroll
        for (int rt = 0; rt < 2; ++rt) {
            int arow = r0 + rt * 16 + nlane;
            #pragma unroll
            for (int kb = 0; kb < 4; ++kb)
                afrag[rt][kb] = load_frag(src + arow * PITCH, kb * 32 + hi_half);
        }

        #pragma unroll
        for (int c = 0; c < 8; ++c) {
            int n0 = c * 16;
            int bn = n0 + nlane;                 // B column owned by this lane
            v8f acc0 = {};
            v8f acc1 = {};
            #pragma unroll
            for (int kb = 0; kb < 4; ++kb) {
                v16h bfrag = load_frag(wbuf + bn * PITCH, kb * 32 + hi_half);
                acc0 = __builtin_amdgcn_wmma_f32_16x16x32_f16(
                           false, afrag[0][kb], false, bfrag,
                           (short)0, acc0, false, false);
                acc1 = __builtin_amdgcn_wmma_f32_16x16x32_f16(
                           false, afrag[1][kb], false, bfrag,
                           (short)0, acc1, false, false);
            }
            // ReLU + repack to f16 activations (C/D layout: VGPR i -> M=i or 8+i)
            int mb0  = r0 + mhalf;
            int ncol = n0 + nlane;
            #pragma unroll
            for (int i = 0; i < 8; ++i) {
                dst[(mb0 + i) * PITCH + ncol]      = (_Float16)fmaxf(acc0[i], 0.f);
                dst[(mb0 + 16 + i) * PITCH + ncol] = (_Float16)fmaxf(acc1[i], 0.f);
            }
        }
        __syncthreads();
        _Float16* t = src; src = dst; dst = t;
    }

    // ---- output head: C = h . Wout, 1 point per thread ----
    {
        float tot = 0.f;
        #pragma unroll 8
        for (int k = 0; k < 128; ++k)
            tot += (float)src[tid * PITCH + k] * Wout[k];
        int gp = wg * MTILE + tid;
        C_out[gp] = tot;
        Cpos[gp]  = fmaxf(tot, 0.f);
    }
}

// ---------------------------------------------------------------------------
// K3: 5x5 box filter (zero-pad SAME) on obj, masked to free cells, + relu(C)
// ---------------------------------------------------------------------------
__global__ void cost_kernel(const float* __restrict__ obj, const float* __restrict__ Cpos,
                            const int* __restrict__ bt, float* __restrict__ Cs) {
    int idx = blockIdx.x * blockDim.x + threadIdx.x;
    if (idx >= NPIX) return;
    int y = idx >> 9, xx = idx & 511;
    float s = 0.f;
    #pragma unroll
    for (int dy = -2; dy <= 2; ++dy) {
        int yy = y + dy;
        if (yy < 0 || yy >= RH) continue;
        #pragma unroll
        for (int dx = -2; dx <= 2; ++dx) {
            int xc = xx + dx;
            if (xc < 0 || xc >= RW) continue;
            s += obj[yy * RW + xc];
        }
    }
    float extra = (bt[idx] == 0) ? s * (1.f / 25.f) : 0.f;
    Cs[idx] = Cpos[idx] + extra;
}

// ---------------------------------------------------------------------------
// K4: one Jacobi sweep (edge-clamped 8-neighbor mean + cost; boundaries pinned)
// ---------------------------------------------------------------------------
__global__ void jacobi_kernel(const float* __restrict__ Vin, float* __restrict__ Vout,
                              const float* __restrict__ Cs, const int* __restrict__ bt,
                              const float* __restrict__ bc) {
    int idx = blockIdx.x * blockDim.x + threadIdx.x;
    if (idx >= NPIX) return;
    if (bt[idx] != 0) { Vout[idx] = bc[idx]; return; }
    int y = idx >> 9, x = idx & 511;
    int ym = y > 0 ? y - 1 : 0, yp = y < RH - 1 ? y + 1 : RH - 1;
    int xm = x > 0 ? x - 1 : 0, xp = x < RW - 1 ? x + 1 : RW - 1;
    float nb = (Vin[ym * RW + xm] + Vin[ym * RW + x] + Vin[ym * RW + xp]
              + Vin[y  * RW + xm]                    + Vin[y  * RW + xp]
              + Vin[yp * RW + xm] + Vin[yp * RW + x] + Vin[yp * RW + xp]) * 0.125f;
    Vout[idx] = nb + Cs[idx];
}

// ---------------------------------------------------------------------------
// K5: bilinear grid sample, align_corners=True, border clamp
// ---------------------------------------------------------------------------
__global__ void sample_kernel(const float* __restrict__ x, const float* __restrict__ V,
                              float* __restrict__ out) {
    int i = blockIdx.x * blockDim.x + threadIdx.x;
    if (i >= NPIX) return;
    float gx = x[2 * i], gy = x[2 * i + 1];
    float ix = (gx + 1.f) * 0.5f * (RW - 1);
    float iy = (gy + 1.f) * 0.5f * (RH - 1);
    ix = fminf(fmaxf(ix, 0.f), (float)(RW - 1));
    iy = fminf(fmaxf(iy, 0.f), (float)(RH - 1));
    int x0 = (int)floorf(ix), y0 = (int)floorf(iy);
    int x1 = x0 + 1 < RW ? x0 + 1 : RW - 1;
    int y1 = y0 + 1 < RH ? y0 + 1 : RH - 1;
    float wx = ix - x0, wy = iy - y0;
    out[i] = V[y0 * RW + x0] * (1.f - wx) * (1.f - wy)
           + V[y0 * RW + x1] * wx * (1.f - wy)
           + V[y1 * RW + x0] * (1.f - wx) * wy
           + V[y1 * RW + x1] * wx * wy;
}

// ---------------------------------------------------------------------------
extern "C" void kernel_launch(void* const* d_in, const int* in_sizes, int n_in,
                              void* d_out, int out_size, void* d_ws, size_t ws_size,
                              hipStream_t stream) {
    (void)in_sizes; (void)n_in; (void)out_size; (void)ws_size;
    const float* x    = (const float*)d_in[0];   // [N,2]
    const int*   bt   = (const int*)  d_in[1];   // [H,W]
    const float* bc   = (const float*)d_in[2];   // [H,W]
    const float* Benc = (const float*)d_in[3];   // [64,2]
    const float* Wh   = (const float*)d_in[4];   // [4,128,128]
    const float* Wout = (const float*)d_in[5];   // [128,1]

    float* out = (float*)d_out;                  // [0,N): sampled out; [N,2N): C

    char* ws = (char*)d_ws;
    _Float16* Wt16 = (_Float16*)ws;                               // 131072 B
    float* Cpos = (float*)(ws + 131072);
    float* obj  = Cpos + NPIX;
    float* Cs   = obj  + NPIX;
    float* Va   = Cs   + NPIX;
    float* Vb   = Va   + NPIX;

    init_kernel <<<NPIX / 256, 256, 0, stream>>>(bt, bc, Va, obj);
    wconv_kernel<<<(4 * 128 * 128) / 256, 256, 0, stream>>>(Wh, Wt16);

    size_t lds_bytes = (size_t)3 * MTILE * PITCH * sizeof(_Float16);   // ~102 KB < 320 KB/WGP
    mlp_kernel<<<NPIX / MTILE, 128, lds_bytes, stream>>>(x, Benc, Wt16, Wout,
                                                         out + NPIX, Cpos);

    cost_kernel<<<NPIX / 256, 256, 0, stream>>>(obj, Cpos, bt, Cs);

    float* a = Va; float* b = Vb;
    for (int it = 0; it < 5; ++it) {
        jacobi_kernel<<<NPIX / 256, 256, 0, stream>>>(a, b, Cs, bt, bc);
        float* t = a; a = b; b = t;
    }
    sample_kernel<<<NPIX / 256, 256, 0, stream>>>(x, a, out);
}
